// MEGNetBlock_74998718922913
// MI455X (gfx1250) — matrix operand order repaired and checked by reading, not compiled, not run on hardware
//
#include <hip/hip_runtime.h>
#include <hip/hip_bf16.h>

// ---------------------------------------------------------------------------
// MEGNet block for MI455X (gfx1250, wave32). All GEMMs run on
// v_wmma_f32_16x16x32_bf16 with f32 accumulation. Double-buffered LDS with
// A-only register pipeline: streaming/gathered A loads for tile k+1 are
// issued before the WMMA burst on tile k (HBM latency hidden in-wave), while
// L2-resident weight tiles (B) are staged load->store in the commit phase
// (transient registers; keeps the wave under the VGPR cap, no scratch
// spills). Concat/gather fused into branch-free 16B-vectorized A staging;
// weights pre-transposed to [N x K] bf16; bf16 packing via v_perm_b32; skip
// connections fused into the final epilogues.
// ---------------------------------------------------------------------------

typedef __attribute__((ext_vector_type(16))) __bf16 v16bf;
typedef __attribute__((ext_vector_type(8)))  __bf16 v8bf;
typedef __attribute__((ext_vector_type(8)))  float  v8f;

static constexpr int NE = 300000;
static constexpr int NN = 20000;
static constexpr int NG = 32;
static constexpr int Dm = 128;
static constexpr int Hm = 256;

__device__ __forceinline__ unsigned pk2bf(float lo, float hi) {
  // round-bias then pack the two high halves with one v_perm_b32
  unsigned ulo = __builtin_bit_cast(unsigned, lo) + 0x8000u;
  unsigned uhi = __builtin_bit_cast(unsigned, hi) + 0x8000u;
  return __builtin_amdgcn_perm(uhi, ulo, 0x07060302u);
}
__device__ __forceinline__ __bf16 tobf(float f) {
  unsigned u = __builtin_bit_cast(unsigned, f);
  unsigned short h = (unsigned short)((u + 0x7FFFu + ((u >> 16) & 1u)) >> 16);
  return __builtin_bit_cast(__bf16, h);
}
__device__ __forceinline__ float softplusf(float x) {
  return fmaxf(x, 0.0f) + __logf(1.0f + __expf(-fabsf(x)));
}

struct GemmArgs {
  const float*  Af32;                       // mode 0
  const __bf16* Abf;                        // mode 1
  const __bf16* catA; const __bf16* catB; const __bf16* catC; // modes 2..4
  const int* src; const int* dst; const int* n2g;
  const __bf16* WT;  const float* bias;     // WT is [N x K] row-major bf16
  __bf16* out_bf; float* out_f32; float* dout; const float* skip;
  int M, K, N;
};

// A-source modes: 0=f32 direct, 1=bf16 direct, 2=edge concat, 3=node concat,
// 4=attr concat.  Tile: BM=128, BN=128, BK=32, 8 waves (256 thr, wave32).
template <int AMODE>
__global__ __launch_bounds__(256)
void gemm_softplus(GemmArgs g) {
  constexpr int AS = 40;                    // padded LDS row stride (bf16)
  __shared__ __align__(16) __bf16 Ash[2][128 * AS];
  __shared__ __align__(16) __bf16 Bsh[2][128 * AS];

  const int t    = threadIdx.x;
  const int lane = t & 31;
  const int wave = t >> 5;
  const int wm   = wave & 3;                // M sub-tile (32 rows each)
  const int wn   = wave >> 2;               // N sub-tile (64 cols each)
  const long m0  = (long)blockIdx.x * 128;
  const int  n0  = blockIdx.y * 128;

  v8f acc[2][4] = {};

  // ---- staging helpers: 2 chunks of 8 bf16 (16B) per thread ---------------
  auto stageA = [&](int k0, uint4 out[2]) {
    const int seg  = k0 >> 7;               // concat segment (uniform per tile)
    const int off0 = k0 & 127;
    #pragma unroll
    for (int i = 0; i < 2; ++i) {
      int chunk = t + i * 256;              // 512 chunks total
      int r  = chunk >> 2;
      int kc = (chunk & 3) * 8;
      long row = m0 + r;
      if (row >= g.M) row = g.M - 1;        // clamp: masked in epilogue
      uint4 bits;
      if (AMODE == 0) {
        const float* p = g.Af32 + row * (long)g.K + (k0 + kc);
        float4 f0 = *(const float4*)(p);
        float4 f1 = *(const float4*)(p + 4);
        bits.x = pk2bf(f0.x, f0.y); bits.y = pk2bf(f0.z, f0.w);
        bits.z = pk2bf(f1.x, f1.y); bits.w = pk2bf(f1.z, f1.w);
      } else if (AMODE == 1) {
        bits = *(const uint4*)(g.Abf + row * (long)g.K + (k0 + kc));
      } else {
        const __bf16* base;
        if (AMODE == 2) {                   // [v[src] | v[dst] | e | u[g(src)]]
          if      (seg == 0) base = g.catA + (long)g.src[row] * 128;
          else if (seg == 1) base = g.catA + (long)g.dst[row] * 128;
          else if (seg == 2) base = g.catB + row * 128;
          else               base = g.catC + (long)g.n2g[g.src[row]] * 128;
        } else if (AMODE == 3) {            // [v | ve | u[g(node)]]
          if      (seg == 0) base = g.catA + row * 128;
          else if (seg == 1) base = g.catB + row * 128;
          else               base = g.catC + (long)g.n2g[row] * 128;
        } else {                            // [u | ue | uv]
          base = (seg == 0 ? g.catA : (seg == 1 ? g.catB : g.catC)) + row * 128;
        }
        bits = *(const uint4*)(base + off0 + kc);
      }
      out[i] = bits;
    }
  };
  auto commitA = [&](int buf, const uint4 aR[2]) {
    #pragma unroll
    for (int i = 0; i < 2; ++i) {
      int chunk = t + i * 256;
      int r  = chunk >> 2;
      int kc = (chunk & 3) * 8;
      *(uint4*)(&Ash[buf][r * AS + kc]) = aR[i];
    }
  };
  // B (weights) is L2-resident: load->store directly, transient registers.
  auto stageBdirect = [&](int buf, int k0) {
    #pragma unroll
    for (int i = 0; i < 2; ++i) {
      int chunk = t + i * 256;
      int n  = chunk >> 2;
      int kc = (chunk & 3) * 8;
      uint4 bits = *(const uint4*)(g.WT + (long)(n0 + n) * g.K + (k0 + kc));
      *(uint4*)(&Bsh[buf][n * AS + kc]) = bits;
    }
  };

  // ---- software pipeline: prologue -----------------------------------------
  uint4 aR[2];
  stageA(0, aR);
  commitA(0, aR);
  stageBdirect(0, 0);
  __syncthreads();

  const int ntiles = g.K >> 5;
  for (int kt = 0; kt < ntiles; ++kt) {
    const int  cur  = kt & 1;
    const bool more = (kt + 1) < ntiles;

    if (more)                               // issue next A tile's global loads
      stageA((kt + 1) * 32, aR);
    if (kt + 2 < ntiles) {                  // prefetch 2 tiles ahead
      long prow = m0 + (t >> 2);
      if (prow >= g.M) prow = g.M - 1;
      int pk = (kt + 2) * 32 + (t & 3) * 8;
      if (AMODE == 0)      __builtin_prefetch(g.Af32 + prow * (long)g.K + pk, 0, 1);
      else if (AMODE == 1) __builtin_prefetch(g.Abf  + prow * (long)g.K + pk, 0, 1);
    }

    // ---- fragments per ISA 16-bit layouts, from LDS buffer `cur` ----------
    const int arow  = wm * 32 + (lane & 15);
    const int khalf = (lane >> 4) * 8;      // lanes 0-15: K0-7/16-23; 16-31: K8-15/24-31
    v16bf afrag[2];
    #pragma unroll
    for (int mt = 0; mt < 2; ++mt) {
      const __bf16* ap = &Ash[cur][(arow + mt * 16) * AS];
      v8bf lo = *(const v8bf*)(ap + khalf);
      v8bf hi = *(const v8bf*)(ap + 16 + khalf);
      afrag[mt] = __builtin_shufflevector(lo, hi, 0,1,2,3,4,5,6,7,8,9,10,11,12,13,14,15);
    }
    const int bcol = wn * 64 + (lane & 15);
    const int kb   = (lane >> 4) * 8;       // lanes 0-15: K0-7/16-23; 16-31: K8-15/24-31
    v16bf bfrag[4];
    #pragma unroll
    for (int nt = 0; nt < 4; ++nt) {
      const __bf16* bp = &Bsh[cur][(bcol + nt * 16) * AS + kb];
      v8bf lo = *(const v8bf*)(bp);
      v8bf hi = *(const v8bf*)(bp + 16);
      bfrag[nt] = __builtin_shufflevector(lo, hi, 0,1,2,3,4,5,6,7,8,9,10,11,12,13,14,15);
    }
    #pragma unroll
    for (int mt = 0; mt < 2; ++mt)
      #pragma unroll
      for (int nt = 0; nt < 4; ++nt)
        acc[mt][nt] = __builtin_amdgcn_wmma_f32_16x16x32_bf16(
            false, afrag[mt], false, bfrag[nt], (short)0, acc[mt][nt],
            false, false);

    if (more) {                             // commit next tile, flip buffers
      commitA(cur ^ 1, aR);
      stageBdirect(cur ^ 1, (kt + 1) * 32);
      __syncthreads();
    }
  }

  // ---- epilogue: bias + softplus (+skip) ---------------------------------
  const int colq = lane & 15;
  const int rsel = (lane >> 4) * 8;         // C/D layout: lanes>=16 hold M=8..15
  float biasv[4];
  #pragma unroll
  for (int nt = 0; nt < 4; ++nt)
    biasv[nt] = g.bias[n0 + wn * 64 + nt * 16 + colq];

  #pragma unroll
  for (int mt = 0; mt < 2; ++mt) {
    #pragma unroll
    for (int nt = 0; nt < 4; ++nt) {
      const int C = n0 + wn * 64 + nt * 16 + colq;
      #pragma unroll
      for (int i = 0; i < 8; ++i) {
        long R = m0 + wm * 32 + mt * 16 + rsel + i;
        if (R < g.M) {
          float a = softplusf(acc[mt][nt][i] + biasv[nt]);
          long o = R * (long)g.N + C;
          if (g.out_bf)  g.out_bf[o]  = tobf(a);
          if (g.out_f32) g.out_f32[o] = a;
          if (g.dout)    g.dout[o]    = a + g.skip[o];
        }
      }
    }
  }
}

// ---------------------------------------------------------------------------
// Small helper kernels
// ---------------------------------------------------------------------------

// weights: f32 [K x N] -> bf16 transposed [N x K]
__global__ void cvt_transpose_k(const float* in, __bf16* out, int K, int N) {
  long i = (long)blockIdx.x * blockDim.x + threadIdx.x;
  if (i < (long)K * N) {
    int k = (int)(i / N), n = (int)(i % N);
    out[(long)n * K + k] = tobf(in[i]);
  }
}

__global__ void zero_f32_k(float* p, long n) {
  long i = (long)blockIdx.x * blockDim.x + threadIdx.x;
  if (i < n) p[i] = 0.0f;
}

// per-edge-element scatter: ve_acc[dst] += e_new ; ue_acc[g(src)] += e_new
__global__ void edge_scatter_k(const float* e_new, const int* dst,
                               const int* src, const int* n2g,
                               float* ve_acc, float* ue_acc, long n) {
  long idx = (long)blockIdx.x * blockDim.x + threadIdx.x;
  if (idx >= n) return;
  long j = idx >> 7; int c = (int)(idx & 127);
  float v = e_new[idx];
  atomicAdd(&ve_acc[(long)dst[j] * 128 + c], v);
  atomicAdd(&ue_acc[(long)n2g[src[j]] * 128 + c], v);
}

__global__ void edge_counts_k(const int* dst, const int* src, const int* n2g,
                              float* ve_cnt, float* ue_cnt, int nE) {
  int j = blockIdx.x * blockDim.x + threadIdx.x;
  if (j < nE) {
    atomicAdd(&ve_cnt[dst[j]], 1.0f);
    atomicAdd(&ue_cnt[n2g[src[j]]], 1.0f);
  }
}

__global__ void node_scatter_k(const float* v_new, const int* n2g,
                               float* uv_acc, long n) {
  long idx = (long)blockIdx.x * blockDim.x + threadIdx.x;
  if (idx >= n) return;
  long i = idx >> 7; int c = (int)(idx & 127);
  atomicAdd(&uv_acc[(long)n2g[i] * 128 + c], v_new[idx]);
}

__global__ void node_counts_k(const int* n2g, float* uv_cnt, int nN) {
  int i = blockIdx.x * blockDim.x + threadIdx.x;
  if (i < nN) atomicAdd(&uv_cnt[n2g[i]], 1.0f);
}

__global__ void seg_mean_fin_k(const float* acc, const float* cnt,
                               __bf16* outb, long n) {
  long i = (long)blockIdx.x * blockDim.x + threadIdx.x;
  if (i < n) {
    long s = i >> 7;
    outb[i] = tobf(acc[i] / fmaxf(cnt[s], 1.0f));
  }
}

// ---------------------------------------------------------------------------
extern "C" void kernel_launch(void* const* d_in, const int* in_sizes, int n_in,
                              void* d_out, int out_size, void* d_ws, size_t ws_size,
                              hipStream_t stream) {
  const float* edge_feat  = (const float*)d_in[0];
  const float* node_feat  = (const float*)d_in[1];
  const float* graph_attr = (const float*)d_in[2];
  const int*   src        = (const int*)d_in[3];
  const int*   dst        = (const int*)d_in[4];
  const int*   n2g        = (const int*)d_in[5];
  const float* pe_W = (const float*)d_in[6];  const float* pe_b = (const float*)d_in[7];
  const float* pn_W = (const float*)d_in[8];  const float* pn_b = (const float*)d_in[9];
  const float* pa_W = (const float*)d_in[10]; const float* pa_b = (const float*)d_in[11];
  const float* eW0 = (const float*)d_in[12];  const float* eb0 = (const float*)d_in[13];
  const float* eW1 = (const float*)d_in[14];  const float* eb1 = (const float*)d_in[15];
  const float* eW2 = (const float*)d_in[16];  const float* eb2 = (const float*)d_in[17];
  const float* nW0 = (const float*)d_in[18];  const float* nb0 = (const float*)d_in[19];
  const float* nW1 = (const float*)d_in[20];  const float* nb1 = (const float*)d_in[21];
  const float* nW2 = (const float*)d_in[22];  const float* nb2 = (const float*)d_in[23];
  const float* aW0 = (const float*)d_in[24];  const float* ab0 = (const float*)d_in[25];
  const float* aW1 = (const float*)d_in[26];  const float* ab1 = (const float*)d_in[27];
  const float* aW2 = (const float*)d_in[28];  const float* ab2 = (const float*)d_in[29];

  float* d_e = (float*)d_out;                       // e_new + e_in  [NE x D]
  float* d_v = d_e + (size_t)NE * Dm;               // v_new + v_in  [NN x D]
  float* d_u = d_v + (size_t)NN * Dm;               // u_new + u_in  [NG x D]

  // ---- workspace bump allocator ------------------------------------------
  char* ws = (char*)d_ws;
  size_t cur = 0;
  auto take = [&](size_t bytes) -> char* {
    char* p = ws + cur;
    cur += (bytes + 255) & ~(size_t)255;
    return p;
  };
  __bf16* wpe = (__bf16*)take((size_t)Dm * Dm * 2);
  __bf16* wpn = (__bf16*)take((size_t)Dm * Dm * 2);
  __bf16* wpa = (__bf16*)take((size_t)Dm * Dm * 2);
  __bf16* we0 = (__bf16*)take((size_t)4 * Dm * Hm * 2);
  __bf16* we1 = (__bf16*)take((size_t)Hm * Hm * 2);
  __bf16* we2 = (__bf16*)take((size_t)Hm * Dm * 2);
  __bf16* wn0 = (__bf16*)take((size_t)3 * Dm * Hm * 2);
  __bf16* wn1 = (__bf16*)take((size_t)Hm * Hm * 2);
  __bf16* wn2 = (__bf16*)take((size_t)Hm * Dm * 2);
  __bf16* wa0 = (__bf16*)take((size_t)3 * Dm * Hm * 2);
  __bf16* wa1 = (__bf16*)take((size_t)Hm * Hm * 2);
  __bf16* wa2 = (__bf16*)take((size_t)Hm * Dm * 2);

  __bf16* e_bf  = (__bf16*)take((size_t)NE * Dm * 2);
  __bf16* v_bf  = (__bf16*)take((size_t)NN * Dm * 2);
  __bf16* u_bf  = (__bf16*)take((size_t)NG * Dm * 2);
  __bf16* h0    = (__bf16*)take((size_t)NE * Hm * 2);
  __bf16* h1    = (__bf16*)take((size_t)NE * Hm * 2);
  float*  e_new = (float*)take((size_t)NE * Dm * 4);
  float*  ve_acc= (float*)take((size_t)NN * Dm * 4);
  float*  ve_cnt= (float*)take((size_t)NN * 4);
  __bf16* ve_bf = (__bf16*)take((size_t)NN * Dm * 2);
  float*  ue_acc= (float*)take((size_t)NG * Dm * 4);
  float*  ue_cnt= (float*)take((size_t)NG * 4);
  __bf16* ue_bf = (__bf16*)take((size_t)NG * Dm * 2);
  __bf16* nh0   = (__bf16*)take((size_t)NN * Hm * 2);
  __bf16* nh1   = (__bf16*)take((size_t)NN * Hm * 2);
  float*  v_new = (float*)take((size_t)NN * Dm * 4);
  float*  uv_acc= (float*)take((size_t)NG * Dm * 4);
  float*  uv_cnt= (float*)take((size_t)NG * 4);
  __bf16* uv_bf = (__bf16*)take((size_t)NG * Dm * 2);
  __bf16* ah0   = (__bf16*)take((size_t)NG * Hm * 2);
  __bf16* ah1   = (__bf16*)take((size_t)NG * Hm * 2);

  auto cvtT = [&](const float* in, __bf16* out, int K, int N) {
    long n = (long)K * N;
    cvt_transpose_k<<<dim3((unsigned)((n + 255) / 256)), 256, 0, stream>>>(in, out, K, N);
  };
  auto zero = [&](float* p, long n) {
    zero_f32_k<<<dim3((unsigned)((n + 255) / 256)), 256, 0, stream>>>(p, n);
  };
  auto gemm = [&](int amode, const float* Af32, const __bf16* Abf,
                  const __bf16* cA, const __bf16* cB, const __bf16* cC,
                  const __bf16* WT, const float* bias,
                  __bf16* outb, float* outf, float* doutp, const float* skip,
                  int M, int K, int N) {
    GemmArgs g{Af32, Abf, cA, cB, cC, src, dst, n2g,
               WT, bias, outb, outf, doutp, skip, M, K, N};
    dim3 grid((M + 127) / 128, N / 128);
    switch (amode) {
      case 0: gemm_softplus<0><<<grid, 256, 0, stream>>>(g); break;
      case 1: gemm_softplus<1><<<grid, 256, 0, stream>>>(g); break;
      case 2: gemm_softplus<2><<<grid, 256, 0, stream>>>(g); break;
      case 3: gemm_softplus<3><<<grid, 256, 0, stream>>>(g); break;
      default: gemm_softplus<4><<<grid, 256, 0, stream>>>(g); break;
    }
  };

  // 1) weights f32 [K x N] -> bf16 transposed [N x K]
  cvtT(pe_W, wpe, Dm, Dm);       cvtT(pn_W, wpn, Dm, Dm);  cvtT(pa_W, wpa, Dm, Dm);
  cvtT(eW0, we0, 4 * Dm, Hm);    cvtT(eW1, we1, Hm, Hm);   cvtT(eW2, we2, Hm, Dm);
  cvtT(nW0, wn0, 3 * Dm, Hm);    cvtT(nW1, wn1, Hm, Hm);   cvtT(nW2, wn2, Hm, Dm);
  cvtT(aW0, wa0, 3 * Dm, Hm);    cvtT(aW1, wa1, Hm, Hm);   cvtT(aW2, wa2, Hm, Dm);

  // 2) zero segment accumulators
  zero(ve_acc, (long)NN * Dm); zero(ve_cnt, NN);
  zero(ue_acc, (long)NG * Dm); zero(ue_cnt, NG);
  zero(uv_acc, (long)NG * Dm); zero(uv_cnt, NG);

  // 3) pre-dense MLPs (softplus) -> bf16 activations
  gemm(0, edge_feat,  nullptr, nullptr, nullptr, nullptr, wpe, pe_b,
       e_bf, nullptr, nullptr, nullptr, NE, Dm, Dm);
  gemm(0, node_feat,  nullptr, nullptr, nullptr, nullptr, wpn, pn_b,
       v_bf, nullptr, nullptr, nullptr, NN, Dm, Dm);
  gemm(0, graph_attr, nullptr, nullptr, nullptr, nullptr, wpa, pa_b,
       u_bf, nullptr, nullptr, nullptr, NG, Dm, Dm);

  // 4) edge update MLP: concat fused into A gather; final layer fuses skip
  gemm(2, nullptr, nullptr, v_bf, e_bf, u_bf, we0, eb0,
       h0, nullptr, nullptr, nullptr, NE, 4 * Dm, Hm);
  gemm(1, nullptr, h0, nullptr, nullptr, nullptr, we1, eb1,
       h1, nullptr, nullptr, nullptr, NE, Hm, Hm);
  gemm(1, nullptr, h1, nullptr, nullptr, nullptr, we2, eb2,
       nullptr, e_new, d_e, edge_feat, NE, Hm, Dm);

  // 5) segment means of e_new: ve (by dst) and ue (by graph of src)
  {
    long n = (long)NE * Dm;
    edge_scatter_k<<<dim3((unsigned)((n + 255) / 256)), 256, 0, stream>>>(
        e_new, dst, src, n2g, ve_acc, ue_acc, n);
    edge_counts_k<<<dim3((NE + 255) / 256), 256, 0, stream>>>(
        dst, src, n2g, ve_cnt, ue_cnt, NE);
    seg_mean_fin_k<<<dim3((unsigned)(((long)NN * Dm + 255) / 256)), 256, 0, stream>>>(
        ve_acc, ve_cnt, ve_bf, (long)NN * Dm);
    seg_mean_fin_k<<<dim3((unsigned)(((long)NG * Dm + 255) / 256)), 256, 0, stream>>>(
        ue_acc, ue_cnt, ue_bf, (long)NG * Dm);
  }

  // 6) node update MLP
  gemm(3, nullptr, nullptr, v_bf, ve_bf, u_bf, wn0, nb0,
       nh0, nullptr, nullptr, nullptr, NN, 3 * Dm, Hm);
  gemm(1, nullptr, nh0, nullptr, nullptr, nullptr, wn1, nb1,
       nh1, nullptr, nullptr, nullptr, NN, Hm, Hm);
  gemm(1, nullptr, nh1, nullptr, nullptr, nullptr, wn2, nb2,
       nullptr, v_new, d_v, node_feat, NN, Hm, Dm);

  // 7) graph readout uv (mean of v_new by graph)
  {
    long n = (long)NN * Dm;
    node_scatter_k<<<dim3((unsigned)((n + 255) / 256)), 256, 0, stream>>>(
        v_new, n2g, uv_acc, n);
    node_counts_k<<<dim3((NN + 255) / 256), 256, 0, stream>>>(n2g, uv_cnt, NN);
    seg_mean_fin_k<<<dim3((unsigned)(((long)NG * Dm + 255) / 256)), 256, 0, stream>>>(
        uv_acc, uv_cnt, uv_bf, (long)NG * Dm);
  }

  // 8) graph attr MLP (tiny M=32, same WMMA path)
  gemm(4, nullptr, nullptr, u_bf, ue_bf, uv_bf, wa0, ab0,
       ah0, nullptr, nullptr, nullptr, NG, 3 * Dm, Hm);
  gemm(1, nullptr, ah0, nullptr, nullptr, nullptr, wa1, ab1,
       ah1, nullptr, nullptr, nullptr, NG, Hm, Hm);
  gemm(1, nullptr, ah1, nullptr, nullptr, nullptr, wa2, ab2,
       nullptr, nullptr, d_u, graph_attr, NG, Hm, Dm);
}
